// CNN_LeNetSym_40089224741233
// MI455X (gfx1250) — matrix operand order repaired and checked by reading, not compiled, not run on hardware
//
#include <hip/hip_runtime.h>
#include <hip/hip_bf16.h>

#define NCLUS 512
#define BATCH 128

typedef __attribute__((ext_vector_type(16))) _Float16 v16h;
typedef __attribute__((ext_vector_type(8)))  float    v8f;

// ---------------------------------------------------------------------------
// Cooperative global -> LDS loader. Prefers the gfx1250 async-to-LDS path
// (ASYNCcnt-tracked, bypasses VGPRs); falls back to load + ds_store.
// Caller must __syncthreads() afterwards.
// ---------------------------------------------------------------------------
__device__ __forceinline__ void coop_load_lds(int* dst_lds, const int* __restrict__ src,
                                              int n, int tid, int nthreads) {
#if defined(__has_builtin) && __has_builtin(__builtin_amdgcn_global_load_async_to_lds_b32)
    typedef __attribute__((address_space(1))) int* gip_t;
    typedef __attribute__((address_space(3))) int* lip_t;
    for (int i = tid; i < n; i += nthreads) {
        __builtin_amdgcn_global_load_async_to_lds_b32((gip_t)(src + i), (lip_t)(dst_lds + i),
                                                      /*offset=*/0, /*cpol=*/0);
    }
#if __has_builtin(__builtin_amdgcn_s_wait_asynccnt)
    __builtin_amdgcn_s_wait_asynccnt(0);
#else
    asm volatile("s_wait_asynccnt 0x0" ::: "memory");
#endif
#else
    for (int i = tid; i < n; i += nthreads) dst_lds[i] = src[i];
#endif
}

// ---------------------------------------------------------------------------
// Strided (LDS-column) sort + symbolic reductions. Each thread owns column
// `a = &lds[tid]` with element k at a[k*stride]; uniform-k accesses across a
// wave hit consecutive LDS banks (conflict-free).
// ---------------------------------------------------------------------------
__device__ __forceinline__ void sort_strided(int* a, int n, int stride) {
    for (int gap = n >> 1; gap > 0; gap >>= 1) {
        for (int i = gap; i < n; ++i) {
            int v = a[i * stride];
            int j = i;
            while (j >= gap && a[(j - gap) * stride] > v) {
                a[j * stride] = a[(j - gap) * stride];
                j -= gap;
            }
            a[j * stride] = v;
        }
    }
}

// Matches _conv_reduce: fold through add_lut with conditional re-insertion of
// the partial sum (bubble right keeps the column sorted).
__device__ __forceinline__ int conv_reduce_strided(int* a, int n, int stride,
                                                   const int* __restrict__ add_lut) {
    int t = a[0];
    for (int j = 1; j < n; ++j) {
        int t2  = add_lut[a[j * stride] * NCLUS + t];
        int jc  = (j + 1 < n) ? (j + 1) : (n - 1);
        int nxt = a[jc * stride];
        if ((j < n - 2) && (t2 > nxt)) {
            int p = jc;
            while (p + 1 < n && a[(p + 1) * stride] < t2) {
                a[p * stride] = a[(p + 1) * stride];
                ++p;
            }
            a[p * stride] = t2;
            t = nxt;
        } else {
            t = t2;
        }
    }
    return t;
}

// ---------------------------------------------------------------------------
// Stage 1: discretize pixels -> cluster ids (centroids cached in LDS)
// ---------------------------------------------------------------------------
__global__ __launch_bounds__(256) void k_discretize(const float* __restrict__ x,
                                                    const float* __restrict__ cent,
                                                    int* __restrict__ out, int total) {
    __shared__ float sc[NCLUS];
    for (int i = threadIdx.x; i < NCLUS; i += blockDim.x) sc[i] = cent[i];
    __syncthreads();
    int tid = blockIdx.x * blockDim.x + threadIdx.x;
    if (tid >= total) return;
    float px = x[tid];
    int   best = 0;
    float bd   = fabsf(px - sc[0]);
    for (int c = 1; c < NCLUS; ++c) {
        float d = fabsf(px - sc[c]);
        if (d < bd) { bd = d; best = c; }     // strict < keeps first index (argmin)
    }
    out[tid] = best;
}

// ---------------------------------------------------------------------------
// Stage 2/3: symbolic conv (k=5, s=2), thread per output element (b,oy,ox,co).
// Sort array lives in LDS (interleaved), weight symbols cached in LDS.
// ---------------------------------------------------------------------------
template <int KK, int S, int CIN, int CO, int BLOCK>
__global__ __launch_bounds__(BLOCK) void k_sym_conv(const int* __restrict__ in, int H, int W,
                                                    int OH, int OW,
                                                    const int* __restrict__ wsym,     // (KK*KK*CIN, CO)
                                                    const int* __restrict__ conv_lut,
                                                    const int* __restrict__ add_lut,
                                                    const int* __restrict__ bias_lut, // (NCLUS, CO)
                                                    const int* __restrict__ relu_lut,
                                                    int* __restrict__ out, int total) {
    constexpr int N = KK * KK * CIN;
    __shared__ int lds_w[N * CO];
    __shared__ int lds_arr[N * BLOCK];

    int tid = threadIdx.x;
    coop_load_lds(lds_w, wsym, N * CO, tid, BLOCK);
    __syncthreads();

    int gid = blockIdx.x * BLOCK + tid;
    if (gid >= total) return;
    __builtin_prefetch(add_lut, 0, 1);   // global_prefetch_b8: warm LUT path

    int co = gid % CO;
    int t1 = gid / CO;
    int ox = t1 % OW; t1 /= OW;
    int oy = t1 % OH;
    int b  = t1 / OH;

    int* a = &lds_arr[tid];              // element k at a[k*BLOCK]
    int  m = 0;
    for (int i = 0; i < KK; ++i)
        for (int j = 0; j < KK; ++j)
            for (int c = 0; c < CIN; ++c) {
                int s = in[((b * H + (S * oy + i)) * W + (S * ox + j)) * CIN + c];
                a[m * BLOCK] = conv_lut[s * NCLUS + lds_w[m * CO + co]];
                ++m;
            }
    sort_strided(a, N, BLOCK);
    int t = conv_reduce_strided(a, N, BLOCK, add_lut);
    out[gid] = relu_lut[bias_lut[t * CO + co]];   // bias then relu (fused)
}

// ---------------------------------------------------------------------------
// Stage 4/5: symbolic FC, thread per output element (b, m); LDS sort column.
// TRANS_IN folds the (0,3,1,2)-transpose+reshape of the conv2 output into the
// gather. CACHE_W optionally stages the whole weight-symbol table in LDS.
// ---------------------------------------------------------------------------
template <int NK, int M_, bool TRANS_IN, bool CACHE_W, int BLOCK>
__global__ __launch_bounds__(BLOCK) void k_sym_fc(const int* __restrict__ in,
                                                  const int* __restrict__ wsym,     // (M_, NK)
                                                  const int* __restrict__ fc_lut,
                                                  const int* __restrict__ add_lut,
                                                  const int* __restrict__ bias_lut, // (NCLUS, M_)
                                                  const int* __restrict__ relu_lut,
                                                  int* __restrict__ out, int total) {
    __shared__ int lds_arr[NK * BLOCK];
    __shared__ int lds_w[CACHE_W ? (M_ * NK) : 1];

    int tid = threadIdx.x;
    if (CACHE_W) {
        coop_load_lds(lds_w, wsym, M_ * NK, tid, BLOCK);
        __syncthreads();
    }
    const int* wrow = CACHE_W ? lds_w : wsym;

    int gid = blockIdx.x * BLOCK + tid;
    if (gid >= total) return;
    __builtin_prefetch(add_lut, 0, 1);

    int m = gid % M_;
    int b = gid / M_;

    int* a = &lds_arr[tid];
    for (int k = 0; k < NK; ++k) {
        int xs;
        if (TRANS_IN) {   // x[b, c, i, j] with c=k/25, i=(k%25)/5, j=k%5 from NHWC conv2 out
            int c = k / 25, rem = k % 25, i = rem / 5, j = rem % 5;
            xs = in[((b * 5 + i) * 5 + j) * 16 + c];
        } else {
            xs = in[b * NK + k];
        }
        a[k * BLOCK] = fc_lut[xs * NCLUS + wrow[m * NK + k]];
    }
    sort_strided(a, NK, BLOCK);
    int t = a[0];
    for (int k = 1; k < NK; ++k) t = add_lut[a[k * BLOCK] * NCLUS + t];  // _fc_reduce fold
    out[gid] = relu_lut[bias_lut[t * M_ + m]];
}

// ---------------------------------------------------------------------------
// Stage 6a: gather centroid feats -> f16 A (128x96, K padded 84->96) and pack
// fc3 weights -> f16 B (96x16, B[k][n] = w[n][k], zero padded)
// ---------------------------------------------------------------------------
__global__ __launch_bounds__(256) void k_fc3_prep(const int* __restrict__ xsym,    // (128,84)
                                                  const float* __restrict__ cent,
                                                  const float* __restrict__ w,     // (10,84)
                                                  _Float16* __restrict__ A,
                                                  _Float16* __restrict__ Bm) {
    int tid = blockIdx.x * blockDim.x + threadIdx.x;
    if (tid < BATCH * 96) {
        int bi = tid / 96, k = tid % 96;
        A[tid] = (k < 84) ? (_Float16)cent[xsym[bi * 84 + k]] : (_Float16)0.0f;
    }
    if (tid < 96 * 16) {
        int k = tid / 16, n = tid % 16;
        Bm[tid] = (k < 84 && n < 10) ? (_Float16)w[n * 84 + k] : (_Float16)0.0f;
    }
}

// ---------------------------------------------------------------------------
// Stage 6b: logits = A(128x96) @ B(96x16) via v_wmma_f32_16x16x32_f16.
// One block of 8 wave32 waves; wave w owns M-rows [16w, 16w+16). 3 K-steps.
// A frag (16x32 f16): lane=M%16 (+hi half selects K-base 8), half h ->
//   K = 16*(h/8) + 8*(lane/16) + (h%8)     [ISA 7.12.2 16-bit A layout]
// B frag (32x16 f16): lane%16 = N, half h -> K same as A
// C/D (16x16 f32, 8 VGPRs): VGPR r -> M = r + 8*(lane/16), N = lane%16
// ---------------------------------------------------------------------------
__global__ __launch_bounds__(256) void k_fc3_wmma(const _Float16* __restrict__ A,
                                                  const _Float16* __restrict__ Bm,
                                                  const float* __restrict__ bias,   // (10,)
                                                  float* __restrict__ logits) {     // (128,16)
    int lane = threadIdx.x & 31;
    int wave = threadIdx.x >> 5;            // 0..7
    int M0   = wave * 16;
    v8f acc = {};
#pragma unroll
    for (int kk = 0; kk < 96; kk += 32) {
        v16h af, bf;
        int mrow = M0 + (lane & 15);
        int kb   = (lane >> 4) * 8;
#pragma unroll
        for (int h = 0; h < 16; ++h) {
            int K  = kk + 16 * (h >> 3) + kb + (h & 7);
            af[h]  = A[mrow * 96 + K];
        }
        int ncol = lane & 15;               // B: lane indexes N
#pragma unroll
        for (int h = 0; h < 16; ++h) bf[h] = Bm[(kk + 16 * (h >> 3) + kb + (h & 7)) * 16 + ncol];
        acc = __builtin_amdgcn_wmma_f32_16x16x32_f16(
            /*neg_a=*/false, af, /*neg_b=*/false, bf,
            /*c_mod=*/(short)0, acc, /*reuse_a=*/false, /*reuse_b=*/false);
    }
#pragma unroll
    for (int r = 0; r < 8; ++r) {
        int M = M0 + r + ((lane >> 4) * 8);
        int N = lane & 15;
        logits[M * 16 + N] = acc[r] + ((N < 10) ? bias[N] : 0.0f);
    }
}

// ---------------------------------------------------------------------------
// Stage 7: row softmax over the 10 valid logits -> d_out (128x10 f32)
// ---------------------------------------------------------------------------
__global__ __launch_bounds__(128) void k_softmax10(const float* __restrict__ logits,
                                                   float* __restrict__ out) {
    int b = blockIdx.x * blockDim.x + threadIdx.x;
    if (b >= BATCH) return;
    float v[10];
    float mx = -3.0e38f;
    for (int n = 0; n < 10; ++n) { v[n] = logits[b * 16 + n]; mx = fmaxf(mx, v[n]); }
    float s = 0.0f;
    for (int n = 0; n < 10; ++n) { v[n] = expf(v[n] - mx); s += v[n]; }
    float inv = 1.0f / s;
    for (int n = 0; n < 10; ++n) out[b * 10 + n] = v[n] * inv;
}

// ---------------------------------------------------------------------------
// Host side
// ---------------------------------------------------------------------------
static inline size_t align256(size_t v) { return (v + 255) & ~size_t(255); }

extern "C" void kernel_launch(void* const* d_in, const int* in_sizes, int n_in,
                              void* d_out, int out_size, void* d_ws, size_t ws_size,
                              hipStream_t stream) {
    (void)in_sizes; (void)n_in; (void)out_size; (void)ws_size;

    const float* x_bat     = (const float*)d_in[0];   // (128,1,32,32)
    const float* cent      = (const float*)d_in[1];   // (512,1)
    const int*   conv_lut  = (const int*)d_in[2];
    const int*   fc_lut    = (const int*)d_in[3];
    const int*   add_lut   = (const int*)d_in[4];
    const int*   relu_lut  = (const int*)d_in[5];
    const int*   c1_bias   = (const int*)d_in[6];     // (512,6)
    const int*   c2_bias   = (const int*)d_in[7];     // (512,16)
    const int*   f1_bias   = (const int*)d_in[8];     // (512,120)
    const int*   f2_bias   = (const int*)d_in[9];     // (512,84)
    const int*   c1f       = (const int*)d_in[10];    // (25,6)
    const int*   c2f       = (const int*)d_in[11];    // (150,16)
    const int*   f1f       = (const int*)d_in[12];    // (120,400)
    const int*   f2f       = (const int*)d_in[13];    // (84,120)
    const float* fc3_w     = (const float*)d_in[14];  // (10,84)
    const float* fc3_b     = (const float*)d_in[15];  // (10,)
    float*       out       = (float*)d_out;           // (128,10)

    // Workspace layout
    char*  ws  = (char*)d_ws;
    size_t off = 0;
    int* s0 = (int*)(ws + off); off = align256(off + size_t(BATCH) * 32 * 32 * 4);      // discretized
    int* s1 = (int*)(ws + off); off = align256(off + size_t(BATCH) * 14 * 14 * 6 * 4);  // conv1 out
    int* s2 = (int*)(ws + off); off = align256(off + size_t(BATCH) * 5 * 5 * 16 * 4);   // conv2 out
    int* s3 = (int*)(ws + off); off = align256(off + size_t(BATCH) * 120 * 4);          // fc1 out
    int* s4 = (int*)(ws + off); off = align256(off + size_t(BATCH) * 84 * 4);           // fc2 out
    _Float16* Af = (_Float16*)(ws + off); off = align256(off + size_t(BATCH) * 96 * 2); // A f16
    _Float16* Bf = (_Float16*)(ws + off); off = align256(off + size_t(96) * 16 * 2);    // B f16
    float* lg = (float*)(ws + off); off = align256(off + size_t(BATCH) * 16 * 4);       // logits

    // 1) discretize: 128*32*32 pixels
    {
        int total = BATCH * 32 * 32;
        k_discretize<<<(total + 255) / 256, 256, 0, stream>>>(x_bat, cent, s0, total);
    }
    // 2) conv1: (128,32,32,1) -> (128,14,14,6), n = 25, LDS 25.6KB + 0.6KB
    {
        constexpr int BLK = 256;
        int total = BATCH * 14 * 14 * 6;
        k_sym_conv<5, 2, 1, 6, BLK><<<(total + BLK - 1) / BLK, BLK, 0, stream>>>(
            s0, 32, 32, 14, 14, c1f, conv_lut, add_lut, c1_bias, relu_lut, s1, total);
    }
    // 3) conv2: (128,14,14,6) -> (128,5,5,16), n = 150, LDS 76.8KB + 9.6KB
    {
        constexpr int BLK = 128;
        int total = BATCH * 5 * 5 * 16;
        k_sym_conv<5, 2, 6, 16, BLK><<<(total + BLK - 1) / BLK, BLK, 0, stream>>>(
            s1, 14, 14, 5, 5, c2f, conv_lut, add_lut, c2_bias, relu_lut, s2, total);
    }
    // 4) fc1: transpose(0,3,1,2)+flatten fused into gather; (128,400)->(128,120)
    //    LDS sort columns: 400*64*4 = 102.4KB
    {
        constexpr int BLK = 64;
        int total = BATCH * 120;
        k_sym_fc<400, 120, true, false, BLK><<<(total + BLK - 1) / BLK, BLK, 0, stream>>>(
            s2, f1f, fc_lut, add_lut, f1_bias, relu_lut, s3, total);
    }
    // 5) fc2: (128,120) -> (128,84); LDS 61.4KB columns + 40.3KB cached weights
    {
        constexpr int BLK = 128;
        int total = BATCH * 84;
        k_sym_fc<120, 84, false, true, BLK><<<(total + BLK - 1) / BLK, BLK, 0, stream>>>(
            s3, f2f, fc_lut, add_lut, f2_bias, relu_lut, s4, total);
    }
    // 6) final dense layer via WMMA (f16 in, f32 accum)
    {
        int total = BATCH * 96;   // >= 96*16
        k_fc3_prep<<<(total + 255) / 256, 256, 0, stream>>>(s4, cent, fc3_w, Af, Bf);
        k_fc3_wmma<<<1, 256, 0, stream>>>(Af, Bf, fc3_b, lg);
    }
    // 7) softmax -> d_out
    k_softmax10<<<1, 128, 0, stream>>>(lg, out);
}
